// LinearNF4_63299228008743
// MI455X (gfx1250) — matrix operand order repaired and checked by reading, not compiled, not run on hardware
//
#include <hip/hip_runtime.h>

// ---- CDNA5 WMMA types -------------------------------------------------------
typedef __attribute__((ext_vector_type(16))) __bf16 v16bf;
typedef __attribute__((ext_vector_type(8)))  float  v8f;

union FragBF {
    uint4 u[2];
    v16bf v;
};

// NF4 quantile code table (bitsandbytes)
__constant__ float NF4_CODE_C[16] = {
    -1.0f, -0.6961928009986877f, -0.5250730514526367f, -0.39491748809814453f,
    -0.28444138169288635f, -0.18477343022823334f, -0.09105003625154495f, 0.0f,
    0.07958029955625534f, 0.16093020141124725f, 0.24611230194568634f, 0.33791524171829224f,
    0.44070982933044434f, 0.5626170039176941f, 0.7229568362236023f, 1.0f
};

// f32 -> bf16 with round-to-nearest-even (bit-level, header independent)
static __device__ __forceinline__ unsigned short f2bf(float f) {
    union { float f; unsigned u; } v; v.f = f;
    unsigned r = v.u + 0x7FFFu + ((v.u >> 16) & 1u);
    return (unsigned short)(r >> 16);
}

#define BM  128
#define BN  128
#define BK  64
#define LDT 72      // padded LDS row stride in halves (144 B -> conflict-free)
#define K_DIM 4096
#define N_DIM 4096

__global__ __launch_bounds__(256) void nf4_wmma_gemm(
    const float* __restrict__ X,      // [M, 4096] f32
    const int*   __restrict__ PW,     // [M_out*4096/2] one byte per int32
    const float* __restrict__ AMAX,   // [M_out*4096/64]
    const float* __restrict__ BIAS,   // [4096]
    float*       __restrict__ OUT)    // [M, 4096]
{
    __shared__ __align__(16) unsigned short sA[BM * LDT]; // x tile, bf16, [m][k]
    __shared__ __align__(16) unsigned short sB[BN * LDT]; // W tile, bf16, [n][k]
    __shared__ unsigned nf4tab[256];                      // byte -> packed bf16 pair

    const int tid = threadIdx.x;
    const int bm  = blockIdx.y * BM;
    const int bn  = blockIdx.x * BN;

    // Build byte -> (code[hi], code[lo]) packed-bf16 table (one entry per thread)
    {
        const int b = tid;
        const float chi = NF4_CODE_C[b >> 4];
        const float clo = NF4_CODE_C[b & 15];
        nf4tab[b] = (unsigned)f2bf(chi) | ((unsigned)f2bf(clo) << 16);
    }
    __syncthreads();

    // Wave decomposition: 8 waves -> 2 (M) x 4 (N); each wave: 64x32 outputs
    const int wave  = tid >> 5;
    const int lane  = tid & 31;
    const int lhalf = lane >> 4;    // 0/1 lane-group
    const int lmod  = lane & 15;
    const int wr    = wave >> 2;    // 0..1 -> M offset wr*64
    const int wc    = wave & 3;     // 0..3 -> N offset wc*32

    v8f acc[4][2];
#pragma unroll
    for (int i = 0; i < 4; ++i)
#pragma unroll
        for (int j = 0; j < 2; ++j)
            acc[i][j] = v8f{0.f, 0.f, 0.f, 0.f, 0.f, 0.f, 0.f, 0.f};

    // A-staging assignment: 16 k-chunks (float4) x 16 rows per pass, 8 passes
    const int a_c  = tid & 15;
    const int a_r0 = tid >> 4;
    // W-staging assignment: 2 threads per n-row, each covers 32 k = 16 packed ints
    const int w_n = tid >> 1;
    const int w_h = tid & 1;
    const long long wrow = (long long)(bn + w_n) * (K_DIM / 2); // packed ints per row

    for (int kt = 0; kt < K_DIM; kt += BK) {
        // ---------------- stage A: 128x64 f32 -> bf16 LDS ----------------
#pragma unroll
        for (int p = 0; p < 8; ++p) {
            const int r = a_r0 + p * 16;
            const float4 f4 =
                *(const float4*)(X + (long long)(bm + r) * K_DIM + kt + a_c * 4);
            const unsigned lo = (unsigned)f2bf(f4.x) | ((unsigned)f2bf(f4.y) << 16);
            const unsigned hi = (unsigned)f2bf(f4.z) | ((unsigned)f2bf(f4.w) << 16);
            *(uint2*)&sA[r * LDT + a_c * 4] = make_uint2(lo, hi);
        }

        // -------- stage B: 128x64 NF4 -> dequant bf16 LDS as [n][k] --------
        {
            const float am = AMAX[(long long)(bn + w_n) * (K_DIM / 64) + (kt >> 6)];
            const unsigned short amb = f2bf(am);
            const unsigned am2 = (unsigned)amb | ((unsigned)amb << 16);
            const int* wp = PW + wrow + (kt >> 1) + w_h * 16;
            unsigned short* srow = &sB[w_n * LDT + w_h * 32];
#pragma unroll
            for (int q = 0; q < 4; ++q) {
                const int4 pk = *(const int4*)(wp + q * 4);
                const int vals[4] = {pk.x, pk.y, pk.z, pk.w};
#pragma unroll
                for (int e = 0; e < 4; ++e) {
                    const unsigned pair = nf4tab[vals[e] & 255];
                    unsigned scaled;
                    // packed bf16 multiply: (code_hi, code_lo) * (absmax, absmax)
                    asm volatile("v_pk_mul_bf16 %0, %1, %2"
                                 : "=v"(scaled) : "v"(pair), "v"(am2));
                    *(unsigned*)&srow[(q * 4 + e) * 2] = scaled;
                }
            }
        }
        __syncthreads();

        // Prefetch next iteration's global data while compute runs
        if (kt + BK < K_DIM) {
            __builtin_prefetch(X + (long long)(bm + a_r0) * K_DIM + (kt + BK) + a_c * 4, 0, 0);
            __builtin_prefetch(PW + wrow + ((kt + BK) >> 1) + w_h * 16, 0, 0);
        }

        // ---------------- compute: 2 k-steps of 32, 8 WMMAs each ----------------
#pragma unroll
        for (int kk = 0; kk < BK; kk += 32) {
            FragBF fa[4], fb[2];
#pragma unroll
            for (int i = 0; i < 4; ++i) {
                // A layout: lanes 0-15 K[kk..kk+7]+[kk+16..kk+23], lanes 16-31 +8
                const uint4* pa =
                    (const uint4*)&sA[(wr * 64 + i * 16 + lmod) * LDT + kk + lhalf * 8];
                fa[i].u[0] = pa[0];      // K kk+o .. kk+o+7
                fa[i].u[1] = pa[2];      // K kk+o+16 .. kk+o+23
            }
#pragma unroll
            for (int j = 0; j < 2; ++j) {
                // B layout: lanes 0-15 K[kk..kk+7]+[kk+16..kk+23], lanes 16-31 +8
                const uint4* pb =
                    (const uint4*)&sB[(wc * 32 + j * 16 + lmod) * LDT + kk + lhalf * 8];
                fb[j].u[0] = pb[0];
                fb[j].u[1] = pb[2];
            }
#pragma unroll
            for (int i = 0; i < 4; ++i)
#pragma unroll
                for (int j = 0; j < 2; ++j)
                    acc[i][j] = __builtin_amdgcn_wmma_f32_16x16x32_bf16(
                        /*neg_a=*/false, fa[i].v, /*neg_b=*/false, fb[j].v,
                        /*c_mod=*/(short)0, acc[i][j],
                        /*reuse_a=*/false, /*reuse_b=*/false);
        }
        __syncthreads();
    }

    // ---------------- writeback: D layout M = v + 8*lhalf, N = lmod ----------------
#pragma unroll
    for (int j = 0; j < 2; ++j) {
        const int n  = bn + wc * 32 + j * 16 + lmod;
        const float bv = BIAS[n];
#pragma unroll
        for (int i = 0; i < 4; ++i) {
            const int m0 = bm + wr * 64 + i * 16 + lhalf * 8;
#pragma unroll
            for (int v = 0; v < 8; ++v)
                OUT[(long long)(m0 + v) * N_DIM + n] = acc[i][j][v] + bv;
        }
    }
}

extern "C" void kernel_launch(void* const* d_in, const int* in_sizes, int n_in,
                              void* d_out, int out_size, void* d_ws, size_t ws_size,
                              hipStream_t stream) {
    const float* x    = (const float*)d_in[0];   // [4,2048,4096] f32
    const int*   pw   = (const int*)d_in[1];     // [4096*4096/2] int32 (one byte each)
    const float* amax = (const float*)d_in[2];   // [4096*4096/64]
    const float* bias = (const float*)d_in[3];   // [4096]
    float* out = (float*)d_out;

    const int M = in_sizes[0] / K_DIM;           // 8192
    dim3 grid(N_DIM / BN, M / BM);               // 32 x 64
    dim3 block(256);
    hipLaunchKernelGGL(nf4_wmma_gemm, grid, block, 0, stream, x, pw, amax, bias, out);
}
